// LSTM_35691178229868
// MI455X (gfx1250) — matrix operand order repaired and checked by reading, not compiled, not run on hardware
//
#include <hip/hip_runtime.h>
#include <stdint.h>

// Problem dimensions (fixed by the reference)
#define B_SZ   2048
#define T_SZ   96
#define I_SZ   64
#define H_SZ   1024
#define G_SZ   4096            // 4*H (gates i,f,m,o)
#define O_SZ   256
#define KTOT   (H_SZ + I_SZ)   // 1088: recurrent K = [h | x_t]
#define KT_N   (KTOT / 32)     // 34 K-tiles for gate GEMM
#define NT_N   (G_SZ / 16)     // 256 N-tiles (gate columns)
#define KTO_N  (H_SZ / 32)     // 32 K-tiles for output proj
#define NTO_N  (O_SZ / 16)     // 16 N-tiles for output proj

typedef __bf16 bf16_t;
typedef bf16_t v16bf __attribute__((ext_vector_type(16)));
typedef float  v8f   __attribute__((ext_vector_type(8)));

// CDNA5 async global->LDS path (probe-guarded; falls back to sync staging)
#if defined(__AMDGCN__) && __has_builtin(__builtin_amdgcn_global_load_async_to_lds_b64) && \
    __has_builtin(__builtin_amdgcn_s_wait_asynccnt)
#define HAVE_ASYNC_LDS 1
typedef int v2i __attribute__((vector_size(8)));
typedef __attribute__((address_space(1))) v2i glb_v2i_t;
typedef __attribute__((address_space(3))) v2i lds_v2i_t;
#else
#define HAVE_ASYNC_LDS 0
#endif

union Frag16 {
  v16bf v;
  uint4 q[2];
};

__device__ __forceinline__ uint16_t to_bf16u(float f) {
  uint32_t u = __float_as_uint(f);
  u += 0x7fffu + ((u >> 16) & 1u);   // round-to-nearest-even
  return (uint16_t)(u >> 16);
}

__device__ __forceinline__ float hash_unif(uint32_t x) {
  x ^= x >> 17; x *= 0xed5ad4bbu;
  x ^= x >> 11; x *= 0xac4c1b51u;
  x ^= x >> 15; x *= 0x31848babu;
  x ^= x >> 14;
  return (float)x * (2.0f / 4294967296.0f) - 1.0f;   // ~U(-1,1)
}

__device__ __forceinline__ float sigmoid_f(float x) {
  return 1.0f / (1.0f + __expf(-x));
}
__device__ __forceinline__ float tanh_f(float x) {
  x = fminf(fmaxf(x, -15.0f), 15.0f);
  float e = __expf(-2.0f * x);
  return (1.0f - e) / (1.0f + e);
}

__device__ __forceinline__ v8f wmma_bf16(const Frag16& a, const Frag16& b, v8f c) {
  // D = A(16x32 bf16) * B(32x16 bf16) + C(16x16 f32)
  return __builtin_amdgcn_wmma_f32_16x16x32_bf16(
      /*neg_a=*/false, a.v, /*neg_b=*/false, b.v,
      /*c_mod=*/(short)0, c, /*reuse_a=*/false, /*reuse_b=*/false);
}

// ---------------------------------------------------------------------------
// Prep kernel: pack concatenated gate weights [Wh | Wx] (gate order i,f,m,o)
// into per-tile WMMA B-fragment layout:
//   tile(nt, kt) -> 512 bf16, lane-major: lane*16 + e
//   B element (k', n) of a 32x16 tile: lane = n + 16*(k' >= 16), e = k' & 15
// ---------------------------------------------------------------------------
__global__ void k_pack_gate_weights(
    const float* __restrict__ Wi_h, const float* __restrict__ Wf_h,
    const float* __restrict__ Wm_h, const float* __restrict__ Wo_h,
    const float* __restrict__ Wi_x, const float* __restrict__ Wf_x,
    const float* __restrict__ Wm_x, const float* __restrict__ Wo_x,
    uint16_t* __restrict__ Wc)
{
  int tid = blockIdx.x * blockDim.x + threadIdx.x;
  if (tid >= NT_N * KT_N * 32) return;
  int lane  = tid & 31;
  int tile  = tid >> 5;
  int kt    = tile % KT_N;
  int nt    = tile / KT_N;
  int n     = lane & 15;
  int khalf = lane >> 4;               // 0: K 0..15, 1: K 16..31 of the tile
  int ncol  = nt * 16 + n;
  int gate  = ncol >> 10;              // 0=i 1=f 2=m 3=o
  int row   = ncol & (H_SZ - 1);
  const float* Wh = (gate == 0) ? Wi_h : (gate == 1) ? Wf_h : (gate == 2) ? Wm_h : Wo_h;
  const float* Wx = (gate == 0) ? Wi_x : (gate == 1) ? Wf_x : (gate == 2) ? Wm_x : Wo_x;

  alignas(16) uint16_t outv[16];
#pragma unroll
  for (int e = 0; e < 16; ++e) {
    int k = kt * 32 + khalf * 16 + e;  // 0..1087
    float w = (k < H_SZ) ? Wh[(size_t)row * H_SZ + k]
                         : Wx[(size_t)row * I_SZ + (k - H_SZ)];
    outv[e] = to_bf16u(w);
  }
  uint16_t* dst = Wc + (size_t)tile * 512 + lane * 16;
  *(uint4*)dst       = *(const uint4*)&outv[0];
  *(uint4*)(dst + 8) = *(const uint4*)&outv[8];
}

// Pack W_out [256 x 1024] into B-fragment tiles (same layout, K = H only).
__global__ void k_pack_wout(const float* __restrict__ W_out, uint16_t* __restrict__ Wo)
{
  int tid = blockIdx.x * blockDim.x + threadIdx.x;
  if (tid >= NTO_N * KTO_N * 32) return;
  int lane  = tid & 31;
  int tile  = tid >> 5;
  int kt    = tile % KTO_N;
  int nt    = tile / KTO_N;
  int n     = lane & 15;
  int khalf = lane >> 4;
  int ncol  = nt * 16 + n;

  alignas(16) uint16_t outv[16];
#pragma unroll
  for (int e = 0; e < 16; ++e) {
    int k = kt * 32 + khalf * 16 + e;
    outv[e] = to_bf16u(W_out[(size_t)ncol * H_SZ + k]);
  }
  uint16_t* dst = Wo + (size_t)tile * 512 + lane * 16;
  *(uint4*)dst       = *(const uint4*)&outv[0];
  *(uint4*)(dst + 8) = *(const uint4*)&outv[8];
}

// Concat biases into bx[4096] (i,f,m,o).
__global__ void k_pack_bias(const float* __restrict__ bi, const float* __restrict__ bfv,
                            const float* __restrict__ bm, const float* __restrict__ bo,
                            float* __restrict__ bx)
{
  int tid = blockIdx.x * blockDim.x + threadIdx.x;
  if (tid >= G_SZ) return;
  int gate = tid >> 10, u = tid & (H_SZ - 1);
  const float* p = (gate == 0) ? bi : (gate == 1) ? bfv : (gate == 2) ? bm : bo;
  bx[tid] = p[u];
}

// Convert seq f32 -> bf16 (same [B][T][I] row-major order).
__global__ void k_pack_seq(const float* __restrict__ seq, uint16_t* __restrict__ sq)
{
  size_t tid = (size_t)blockIdx.x * blockDim.x + threadIdx.x;
  if (tid < (size_t)B_SZ * T_SZ * I_SZ) sq[tid] = to_bf16u(seq[tid]);
}

// ---------------------------------------------------------------------------
// Persistent LSTM kernel: each workgroup owns 16 batch rows for all 96 steps.
// 256 threads = 8 waves; wave w handles unit-tiles {w, w+8, ..., w+56}.
// For each unit-tile it accumulates all 4 gate pre-activations via WMMA over
// K = [h(1024) | x_t(64)] = 34 K-tiles, then applies the cell update with
// c held in VGPRs, writing h_next (bf16) to the ping-pong LDS buffer.
// Ends with the fused output projection (K=1024, N=256) into d_out (f32).
// ---------------------------------------------------------------------------
__global__ __launch_bounds__(256)
void k_lstm_persist(const uint16_t* __restrict__ Wc,   // packed gate weights
                    const uint16_t* __restrict__ Wo,   // packed W_out
                    const float* __restrict__ bx,      // concat gate bias [4096]
                    const uint16_t* __restrict__ sq,   // seq bf16 [B][T][I]
                    const float* __restrict__ b_out,   // [256]
                    float* __restrict__ out)           // [B][O] f32
{
  __shared__ uint16_t hx[2][16][KTOT];   // ping-pong [h | x_t], 69,632 B

  const int tid   = threadIdx.x;
  const int lane  = tid & 31;
  const int wave  = tid >> 5;
  const int btile = blockIdx.x;          // 0..127
  const int row16 = lane & 15;           // A-frag row / accumulator column
  const int half  = lane >> 4;           // 0 or 1

  // ---- deterministic pseudo-random h0 (LDS) and c0 (VGPRs) ----
  for (int i = tid; i < 16 * H_SZ; i += 256) {
    int m = i >> 10, u = i & (H_SZ - 1);
    hx[0][m][u] = to_bf16u(hash_unif((uint32_t)(((btile * 16 + m) * H_SZ + u) * 2 + 1)));
  }
  float creg[8][8];
#pragma unroll
  for (int uti = 0; uti < 8; ++uti) {
    int ut = wave + uti * 8;
#pragma unroll
    for (int r = 0; r < 8; ++r) {
      int m = r + 8 * half;
      int u = ut * 16 + row16;
      creg[uti][r] = hash_unif((uint32_t)(((btile * 16 + m) * H_SZ + u) * 2 + 2));
    }
  }

  // ---- hoist t-invariant gate biases into VGPRs ----
  float breg0[8], breg1[8], breg2[8], breg3[8];
#pragma unroll
  for (int uti = 0; uti < 8; ++uti) {
    int ut = wave + uti * 8;
    breg0[uti] = bx[0 * H_SZ + ut * 16 + row16];
    breg1[uti] = bx[1 * H_SZ + ut * 16 + row16];
    breg2[uti] = bx[2 * H_SZ + ut * 16 + row16];
    breg3[uti] = bx[3 * H_SZ + ut * 16 + row16];
  }
  __syncthreads();

  for (int t = 0; t < T_SZ; ++t) {
    const int cur = t & 1, nxt = cur ^ 1;

    // Stage x_t (16 rows x 64 cols bf16) into hx[cur][:, 1024..1087]
    {
      int m  = tid >> 4;           // 0..15
      int i0 = (tid & 15) * 4;     // 0..60
      const uint16_t* src = sq + (((size_t)(btile * 16 + m) * T_SZ + t) * I_SZ + i0);
      uint16_t* dst = &hx[cur][m][H_SZ + i0];
#if HAVE_ASYNC_LDS
      // CDNA5 async DMA: memory -> LDS without VGPR round-trip (ASYNCcnt).
      // AS3 pointer = low 32 bits of the flat LDS address (ISA aperture rule).
      __builtin_amdgcn_global_load_async_to_lds_b64(
          (glb_v2i_t*)(uintptr_t)src,
          (lds_v2i_t*)(uint32_t)(uintptr_t)dst, 0, 0);
      __builtin_amdgcn_s_wait_asynccnt(0);
#else
      *(uint2*)dst = *(const uint2*)src;
#endif
    }
    __syncthreads();

    const uint16_t* hrow = &hx[cur][row16][0];

#pragma unroll
    for (int uti = 0; uti < 8; ++uti) {
      const int ut = wave + uti * 8;

      v8f acc0, acc1, acc2, acc3;
#pragma unroll
      for (int r = 0; r < 8; ++r) {
        acc0[r] = breg0[uti]; acc1[r] = breg1[uti];
        acc2[r] = breg2[uti]; acc3[r] = breg3[uti];
      }

      const int nt0 = 0 * 64 + ut, nt1 = 1 * 64 + ut, nt2 = 2 * 64 + ut, nt3 = 3 * 64 + ut;

      for (int kt = 0; kt < KT_N; ++kt) {
        // A fragment: rows of [h|x] from LDS (two ds_load_b128 per lane)
        Frag16 af;
        const int k0 = kt * 32 + half * 16;
        af.q[0] = *(const uint4*)&hrow[k0];
        af.q[1] = *(const uint4*)&hrow[k0 + 8];

        // B fragments for the 4 gates (L2-resident packed weights)
        const uint16_t* bp0 = Wc + (((size_t)(nt0 * KT_N + kt)) << 9) + lane * 16;
        const uint16_t* bp1 = Wc + (((size_t)(nt1 * KT_N + kt)) << 9) + lane * 16;
        const uint16_t* bp2 = Wc + (((size_t)(nt2 * KT_N + kt)) << 9) + lane * 16;
        const uint16_t* bp3 = Wc + (((size_t)(nt3 * KT_N + kt)) << 9) + lane * 16;
        if (kt + 1 < KT_N)  // pull next tile into WGP$ (locality 3 -> near scope)
          __builtin_prefetch(Wc + (((size_t)(nt0 * KT_N + kt + 1)) << 9) + lane * 16, 0, 3);

        Frag16 bf0, bf1, bf2, bf3;
        bf0.q[0] = *(const uint4*)bp0; bf0.q[1] = *(const uint4*)(bp0 + 8);
        bf1.q[0] = *(const uint4*)bp1; bf1.q[1] = *(const uint4*)(bp1 + 8);
        bf2.q[0] = *(const uint4*)bp2; bf2.q[1] = *(const uint4*)(bp2 + 8);
        bf3.q[0] = *(const uint4*)bp3; bf3.q[1] = *(const uint4*)(bp3 + 8);

        acc0 = wmma_bf16(af, bf0, acc0);
        acc1 = wmma_bf16(af, bf1, acc1);
        acc2 = wmma_bf16(af, bf2, acc2);
        acc3 = wmma_bf16(af, bf3, acc3);
      }

      // Cell update on the accumulators; c never leaves VGPRs.
#pragma unroll
      for (int r = 0; r < 8; ++r) {
        float is = sigmoid_f(acc0[r]);
        float fs = sigmoid_f(acc1[r]);
        float ms = tanh_f(acc2[r]);
        float os = sigmoid_f(acc3[r]);
        float cn = fs * creg[uti][r] + is * ms;
        creg[uti][r] = cn;
        float hn = os * tanh_f(cn);
        hx[nxt][r + 8 * half][ut * 16 + row16] = to_bf16u(hn);
      }
    }
    __syncthreads();
  }

  // ---- fused output projection: out = h_last @ W_out.T + b_out ----
  // h_last lives in hx[0] (T_SZ is even).
  const uint16_t* hrow = &hx[0][row16][0];
#pragma unroll
  for (int nti = 0; nti < 2; ++nti) {
    const int nt = wave * 2 + nti;     // 0..15
    v8f acc;
    {
      float bb = b_out[nt * 16 + row16];
#pragma unroll
      for (int r = 0; r < 8; ++r) acc[r] = bb;
    }
    for (int kt = 0; kt < KTO_N; ++kt) {
      Frag16 af;
      const int k0 = kt * 32 + half * 16;
      af.q[0] = *(const uint4*)&hrow[k0];
      af.q[1] = *(const uint4*)&hrow[k0 + 8];
      const uint16_t* bp = Wo + (((size_t)(nt * KTO_N + kt)) << 9) + lane * 16;
      Frag16 bfr;
      bfr.q[0] = *(const uint4*)bp; bfr.q[1] = *(const uint4*)(bp + 8);
      acc = wmma_bf16(af, bfr, acc);
    }
#pragma unroll
    for (int r = 0; r < 8; ++r)
      out[(size_t)(btile * 16 + r + 8 * half) * O_SZ + nt * 16 + row16] = acc[r];
  }
}

// ---------------------------------------------------------------------------
extern "C" void kernel_launch(void* const* d_in, const int* in_sizes, int n_in,
                              void* d_out, int out_size, void* d_ws, size_t ws_size,
                              hipStream_t stream) {
  (void)in_sizes; (void)n_in; (void)out_size; (void)ws_size;

  // setup_inputs order:
  const float* seq   = (const float*)d_in[0];
  const float* Wf_x  = (const float*)d_in[1];
  const float* bfv   = (const float*)d_in[2];
  const float* Wf_h  = (const float*)d_in[3];
  const float* Wi_x  = (const float*)d_in[4];
  const float* bi    = (const float*)d_in[5];
  const float* Wi_h  = (const float*)d_in[6];
  const float* Wm_x  = (const float*)d_in[7];
  const float* bm    = (const float*)d_in[8];
  const float* Wm_h  = (const float*)d_in[9];
  const float* Wo_x  = (const float*)d_in[10];
  const float* bo    = (const float*)d_in[11];
  const float* Wo_h  = (const float*)d_in[12];
  const float* W_out = (const float*)d_in[13];
  const float* b_out = (const float*)d_in[14];

  // Workspace layout (all offsets 256-byte aligned)
  uint8_t* w = (uint8_t*)d_ws;
  const size_t wc_bytes  = (size_t)NT_N * KT_N * 512 * 2;   //  8,912,896
  const size_t wo_bytes  = (size_t)NTO_N * KTO_N * 512 * 2; //    524,288
  const size_t bx_bytes  = (size_t)G_SZ * 4;                //     16,384
  uint16_t* Wc  = (uint16_t*)w;
  uint16_t* Wop = (uint16_t*)(w + wc_bytes);
  float*    bx  = (float*)   (w + wc_bytes + wo_bytes);
  uint16_t* sq  = (uint16_t*)(w + wc_bytes + wo_bytes + bx_bytes);

  // Prep
  {
    int total = NT_N * KT_N * 32;
    k_pack_gate_weights<<<(total + 255) / 256, 256, 0, stream>>>(
        Wi_h, Wf_h, Wm_h, Wo_h, Wi_x, Wf_x, Wm_x, Wo_x, Wc);
  }
  {
    int total = NTO_N * KTO_N * 32;
    k_pack_wout<<<(total + 255) / 256, 256, 0, stream>>>(W_out, Wop);
  }
  k_pack_bias<<<(G_SZ + 255) / 256, 256, 0, stream>>>(bi, bfv, bm, bo, bx);
  {
    size_t total = (size_t)B_SZ * T_SZ * I_SZ;
    k_pack_seq<<<(unsigned)((total + 255) / 256), 256, 0, stream>>>(seq, sq);
  }

  // Recurrence + fused output projection: 128 WGs x 256 threads (8 waves)
  k_lstm_persist<<<B_SZ / 16, 256, 0, stream>>>(Wc, Wop, bx, sq, b_out, (float*)d_out);
}
